// VLBertEmbeddings_16063177687405
// MI455X (gfx1250) — matrix operand order repaired and checked by reading, not compiled, not run on hardware
//
#include <hip/hip_runtime.h>
#include <hip/hip_bf16.h>

// ---------------- problem constants ----------------
#define BB 64
#define KK 100
#define SS 128
#define HH 768
#define VF 2048
#define CDIM 256
#define FDIM 4096          // 2*VF
#define EPS 1e-12f

typedef __attribute__((ext_vector_type(16))) __bf16 v16bf;
typedef __attribute__((ext_vector_type(8)))  __bf16 v8bf;
typedef __attribute__((ext_vector_type(8)))  float  v8f;

union V16U { v16bf v; v8bf h[2]; };

__device__ __forceinline__ unsigned short f2bf(float f) {
    unsigned int u = __float_as_uint(f);
    unsigned int r = (u + 0x7FFFu + ((u >> 16) & 1u)) >> 16;   // RNE
    return (unsigned short)r;
}

// ---------------- K_w: W_down f32 -> bf16 ----------------
__global__ __launch_bounds__(256) void k_convW(const float* __restrict__ w,
                                               unsigned short* __restrict__ wbf,
                                               int n) {
    int i = blockIdx.x * 256 + threadIdx.x;
    if (i < n) wbf[i] = f2bf(w[i]);
}

// ---------------- K0: text_end[b] = sum(token_ids != 0) ----------------
__global__ __launch_bounds__(128) void k_textend(const int* __restrict__ tok,
                                                 int* __restrict__ tend) {
    __shared__ int s[128];
    int b = blockIdx.x, t = threadIdx.x;
    s[t] = (tok[b * SS + t] != 0) ? 1 : 0;
    __syncthreads();
    for (int off = 64; off > 0; off >>= 1) {
        if (t < off) s[t] += s[t + off];
        __syncthreads();
    }
    if (t == 0) tend[b] = s[0];
}

// ---------------- K1: build bf16 feats [6400, 4096] + mvrc mask ----------------
__global__ __launch_bounds__(256) void k_feats(const float* __restrict__ img,      // [B,K,VF]
                                               const float* __restrict__ loc,      // [B,K,4]
                                               const float* __restrict__ mask_vis, // [VF]
                                               unsigned short* __restrict__ Abf,   // [6400,4096]
                                               unsigned char* __restrict__ mask) {
    __shared__ float red[256];
    __shared__ float spos[4];
    __shared__ int smask;
    int row = blockIdx.x;              // b*K + k
    int t = threadIdx.x;
    const float* irow = img + (long)row * VF;

    // ---- all-zero detection over 2048 elems ----
    float acc = 0.f;
    #pragma unroll
    for (int i = 0; i < 8; ++i) acc += fabsf(irow[t + 256 * i]);
    red[t] = acc;
    __syncthreads();
    for (int off = 128; off > 0; off >>= 1) {
        if (t < off) red[t] += red[t + off];
        __syncthreads();
    }
    if (t == 0) {
        smask = (red[0] == 0.f) ? 1 : 0;
        mask[row] = (unsigned char)smask;
        const float* bx = loc + (long)row * 4;
        float x1 = bx[0], y1 = bx[1], x2 = bx[2], y2 = bx[3];
        spos[0] = (x1 + x2) * 0.5f * 100.f;   // cx
        spos[1] = (y1 + y2) * 0.5f * 100.f;   // cy
        spos[2] = (x2 - x1) * 100.f;          // w
        spos[3] = (y2 - y1) * 100.f;          // h
    }
    __syncthreads();
    int m = smask;
    unsigned short* arow = Abf + (long)row * FDIM;

    // ---- coordinate embeddings: d = t in [0,256) ----
    // inv_dim = 1000^{-d/256}
    float inv = exp2f(-((float)t / (float)CDIM) * 9.965784284662087f); // log2(1000)
    #pragma unroll
    for (int p = 0; p < 4; ++p) {
        float ang = spos[p] * inv;
        float sv, cv;
        __sincosf(ang, &sv, &cv);
        arow[p * (2 * CDIM) + t]        = f2bf(sv);
        arow[p * (2 * CDIM) + CDIM + t] = f2bf(cv);
    }
    // ---- image features (masked rows -> mask_vis embedding) ----
    #pragma unroll
    for (int i = 0; i < 8; ++i) {
        int c = t + 256 * i;
        float v = m ? mask_vis[c] : irow[c];
        arow[VF + c] = f2bf(v);
    }
}

// ---------------- K2: bf16 WMMA GEMM + bias + ReLU ----------------
// C[6400,768] = relu(A[6400,4096] * W^T + b),  W row n holds 4096 contiguous K vals.
// Block: 128 threads = 4 waves. Block tile 64(M) x 128(N); wave tile 64 x 32.
__global__ __launch_bounds__(128) void k_gemm(const unsigned short* __restrict__ Abf,
                                              const unsigned short* __restrict__ Wbf,
                                              const float* __restrict__ bias,
                                              float* __restrict__ ff) {
    int wave = threadIdx.x >> 5;
    int lane = threadIdx.x & 31;
    int m0 = blockIdx.x * 64;
    int n0 = blockIdx.y * 128 + wave * 32;
    int l15 = lane & 15;
    int hi  = lane >> 4;          // 0 for lanes 0-15, 1 for lanes 16-31

    v8f acc[4][2];
    #pragma unroll
    for (int mi = 0; mi < 4; ++mi)
        #pragma unroll
        for (int ni = 0; ni < 2; ++ni)
            acc[mi][ni] = (v8f)0.f;

    // per-lane base pointers (element units)
    const unsigned short* arow[4];
    #pragma unroll
    for (int mi = 0; mi < 4; ++mi)
        arow[mi] = Abf + (long)(m0 + mi * 16 + l15) * FDIM + (hi ? 8 : 0);
    const unsigned short* brow[2];
    #pragma unroll
    for (int ni = 0; ni < 2; ++ni)
        brow[ni] = Wbf + (long)(n0 + ni * 16 + l15) * FDIM + (hi ? 8 : 0);

    for (int k0 = 0; k0 < FDIM; k0 += 32) {
        V16U a[4], bm[2];
        #pragma unroll
        for (int mi = 0; mi < 4; ++mi) {
            a[mi].h[0] = *(const v8bf*)(arow[mi] + k0);        // K slots 0..7 (or 8..15)
            a[mi].h[1] = *(const v8bf*)(arow[mi] + k0 + 16);   // K slots 16..23 (or 24..31)
        }
        #pragma unroll
        for (int ni = 0; ni < 2; ++ni) {
            bm[ni].h[0] = *(const v8bf*)(brow[ni] + k0);
            bm[ni].h[1] = *(const v8bf*)(brow[ni] + k0 + 16);
        }
        #pragma unroll
        for (int mi = 0; mi < 4; ++mi)
            #pragma unroll
            for (int ni = 0; ni < 2; ++ni)
                acc[mi][ni] = __builtin_amdgcn_wmma_f32_16x16x32_bf16(
                    false, a[mi].v, false, bm[ni].v,
                    (short)0, acc[mi][ni], false, false);
    }

    // epilogue: D layout -> lane l15 = column, c[j] = rows j (+8 for upper lanes)
    #pragma unroll
    for (int ni = 0; ni < 2; ++ni) {
        int n = n0 + ni * 16 + l15;
        float bz = bias[n];
        #pragma unroll
        for (int mi = 0; mi < 4; ++mi) {
            #pragma unroll
            for (int j = 0; j < 8; ++j) {
                int mrow = m0 + mi * 16 + j + hi * 8;
                float v = acc[mi][ni][j] + bz;
                ff[(long)mrow * HH + n] = v > 0.f ? v : 0.f;
            }
        }
    }
}

// ---------------- shared 2-value block reduction (blockDim = 256) ----------------
__device__ __forceinline__ void reduce2(float& a, float& b, float* s1, float* s2) {
    int t = threadIdx.x;
    s1[t] = a; s2[t] = b;
    __syncthreads();
    for (int off = 128; off > 0; off >>= 1) {
        if (t < off) { s1[t] += s1[t + off]; s2[t] += s2[t + off]; }
        __syncthreads();
    }
    a = s1[0]; b = s2[0];
    __syncthreads();
}

// ---------------- K3a: object rows ----------------
__global__ __launch_bounds__(256) void k_objrows(const float* __restrict__ ff,
                                                 const unsigned char* __restrict__ mask,
                                                 const int* __restrict__ tend,
                                                 const float* __restrict__ obj_ling_w,
                                                 const float* __restrict__ obj_mask_word_w,
                                                 const float* __restrict__ end_w,
                                                 const float* __restrict__ pos_emb,
                                                 const float* __restrict__ type_emb,
                                                 const float* __restrict__ vw,
                                                 const float* __restrict__ vb,
                                                 const float* __restrict__ lnw,
                                                 const float* __restrict__ lnb,
                                                 float* __restrict__ out) {
    __shared__ float s1[256], s2[256];
    int row = blockIdx.x;                 // b*K + k
    int b = row / KK, k = row % KK;
    int t = threadIdx.x;
    bool last = (k == KK - 1);
    int msk = mask[row];
    int pid = tend[b] + (last ? 1 : 0);

    float fv[3], x[3];
    float sa = 0.f, sb = 0.f;
    #pragma unroll
    for (int i = 0; i < 3; ++i) {
        int j = t + 256 * i;
        fv[i] = ff[(long)row * HH + j];
        sa += fv[i]; sb += fv[i] * fv[i];
    }
    reduce2(sa, sb, s1, s2);
    float m1 = sa / (float)HH;
    float iv1 = rsqrtf(sb / (float)HH - m1 * m1 + EPS);

    sa = 0.f; sb = 0.f;
    #pragma unroll
    for (int i = 0; i < 3; ++i) {
        int j = t + 256 * i;
        float vis = (fv[i] - m1) * iv1 * vw[j] + vb[j];
        float lg = last ? end_w[j] : (msk ? obj_mask_word_w[j] : obj_ling_w[j]);
        x[i] = lg + vis + pos_emb[(long)pid * HH + j] + type_emb[2 * HH + j];
        sa += x[i]; sb += x[i] * x[i];
    }
    reduce2(sa, sb, s1, s2);
    float m2 = sa / (float)HH;
    float iv2 = rsqrtf(sb / (float)HH - m2 * m2 + EPS);
    long base = (long)BB * SS * HH + (long)row * HH;
    #pragma unroll
    for (int i = 0; i < 3; ++i) {
        int j = t + 256 * i;
        out[base + j] = (x[i] - m2) * iv2 * lnw[j] + lnb[j];
    }
}

// ---------------- K3b: text rows ----------------
__global__ __launch_bounds__(256) void k_txtrows(const float* __restrict__ ff,
                                                 const int* __restrict__ tok,
                                                 const int* __restrict__ ttype,
                                                 const int* __restrict__ tend,
                                                 const float* __restrict__ word_emb,
                                                 const float* __restrict__ pos_emb,
                                                 const float* __restrict__ type_emb,
                                                 const float* __restrict__ vw,
                                                 const float* __restrict__ vb,
                                                 const float* __restrict__ lnw,
                                                 const float* __restrict__ lnb,
                                                 float* __restrict__ out) {
    __shared__ float s1[256], s2[256];
    int row = blockIdx.x;                 // b*S + s
    int b = row / SS, s = row % SS;
    int t = threadIdx.x;
    int tk = tok[row];
    int tt = ttype[row];
    int te = tend[b];
    int pid = (s >= te) ? (s + KK) : s;
    long ffrow = (long)(b * KK + KK - 1) * HH;

    float fv[3], x[3];
    float sa = 0.f, sb = 0.f;
    #pragma unroll
    for (int i = 0; i < 3; ++i) {
        int j = t + 256 * i;
        fv[i] = ff[ffrow + j];
        sa += fv[i]; sb += fv[i] * fv[i];
    }
    reduce2(sa, sb, s1, s2);
    float m1 = sa / (float)HH;
    float iv1 = rsqrtf(sb / (float)HH - m1 * m1 + EPS);

    sa = 0.f; sb = 0.f;
    #pragma unroll
    for (int i = 0; i < 3; ++i) {
        int j = t + 256 * i;
        float tvis = (fv[i] - m1) * iv1 * vw[j] + vb[j];
        x[i] = word_emb[(long)tk * HH + j] + tvis
             + pos_emb[(long)pid * HH + j] + type_emb[(long)tt * HH + j];
        sa += x[i]; sb += x[i] * x[i];
    }
    reduce2(sa, sb, s1, s2);
    float m2 = sa / (float)HH;
    float iv2 = rsqrtf(sb / (float)HH - m2 * m2 + EPS);
    long base = (long)row * HH;
    #pragma unroll
    for (int i = 0; i < 3; ++i) {
        int j = t + 256 * i;
        out[base + j] = (x[i] - m2) * iv2 * lnw[j] + lnb[j];
    }
}

// ---------------- launcher ----------------
extern "C" void kernel_launch(void* const* d_in, const int* in_sizes, int n_in,
                              void* d_out, int out_size, void* d_ws, size_t ws_size,
                              hipStream_t stream) {
    const int*   token_ids       = (const int*)d_in[0];
    const float* image_feat      = (const float*)d_in[1];
    const float* image_loc       = (const float*)d_in[2];
    const int*   token_type_ids  = (const int*)d_in[3];
    const float* W_down          = (const float*)d_in[4];
    const float* b_down          = (const float*)d_in[5];
    const float* obj_ling_w      = (const float*)d_in[6];
    const float* obj_mask_word_w = (const float*)d_in[7];
    const float* obj_mask_vis_w  = (const float*)d_in[8];
    const float* end_w           = (const float*)d_in[9];
    const float* word_emb        = (const float*)d_in[10];
    const float* pos_emb         = (const float*)d_in[11];
    const float* type_emb        = (const float*)d_in[12];
    const float* vln_obj_w       = (const float*)d_in[13];
    const float* vln_obj_b       = (const float*)d_in[14];
    const float* vln_txt_w       = (const float*)d_in[15];
    const float* vln_txt_b       = (const float*)d_in[16];
    const float* ln_w            = (const float*)d_in[17];
    const float* ln_b            = (const float*)d_in[18];
    float* out = (float*)d_out;

    // workspace layout (bytes)
    char* ws = (char*)d_ws;
    const long M = (long)BB * KK;                       // 6400
    unsigned short* Abf  = (unsigned short*)(ws);                              // 52,428,800 B
    unsigned short* Wbf  = (unsigned short*)(ws + 52428800);                   //  6,291,456 B
    float*          ffv  = (float*)(ws + 58720256);                            // 19,660,800 B
    unsigned char*  mask = (unsigned char*)(ws + 78381056);                    //      6,400 B
    int*            tend = (int*)(ws + 78387456);                              //        256 B

    // 1) W -> bf16
    {
        int n = HH * FDIM;
        k_convW<<<(n + 255) / 256, 256, 0, stream>>>(W_down, Wbf, n);
    }
    // 2) text_end
    k_textend<<<BB, 128, 0, stream>>>(token_ids, tend);
    // 3) feats (coord sin/cos + masked image) -> bf16
    k_feats<<<(int)M, 256, 0, stream>>>(image_feat, image_loc, obj_mask_vis_w, Abf, mask);
    // 4) WMMA GEMM + bias + ReLU
    {
        dim3 grid((int)(M / 64), HH / 128);   // 100 x 6
        k_gemm<<<grid, 128, 0, stream>>>(Abf, Wbf, b_down, ffv);
    }
    // 5) object rows -> out[B*S*H ..]
    k_objrows<<<(int)M, 256, 0, stream>>>(ffv, mask, tend, obj_ling_w, obj_mask_word_w,
                                          end_w, pos_emb, type_emb,
                                          vln_obj_w, vln_obj_b, ln_w, ln_b, out);
    // 6) text rows -> out[0 ..]
    k_txtrows<<<BB * SS, 256, 0, stream>>>(ffv, token_ids, token_type_ids, tend,
                                           word_emb, pos_emb, type_emb,
                                           vln_txt_w, vln_txt_b, ln_w, ln_b, out);
}
